// SpatialSTEncoder_51823075393592
// MI455X (gfx1250) — compile-verified, matrix operand order checked by reading, not run on hardware
//
#include <hip/hip_runtime.h>

// ---------------------------------------------------------------------------
// Model constants (match reference)
// ---------------------------------------------------------------------------
#define Bsz   32
#define Kg    2048
#define Ntok  2049          // K + 1 (spatial token)
#define Dm    512
#define Hh    8
#define DH    64
#define Mf    256           // random features
#define Ll    4
#define NBb   5
#define FF    2048          // 4*D

typedef unsigned short u16;

// ---------------------------------------------------------------------------
// bf16 helpers (bit-level, storage = unsigned short)
// ---------------------------------------------------------------------------
__device__ __forceinline__ u16 f2bf(float f) {
    union { float f; unsigned u; } x; x.f = f;
    unsigned r = x.u + 0x7FFFu + ((x.u >> 16) & 1u);
    return (u16)(r >> 16);
}
__device__ __forceinline__ float bf2f(u16 h) {
    union { unsigned u; float f; } x; x.u = ((unsigned)h) << 16;
    return x.f;
}
__device__ __forceinline__ float gelu_exact(float v) {
    return 0.5f * v * (1.0f + erff(v * 0.70710678118654752f));
}

// ---------------------------------------------------------------------------
// WMMA types
// ---------------------------------------------------------------------------
typedef __bf16 bf16x16 __attribute__((ext_vector_type(16)));
typedef float  f32x8   __attribute__((ext_vector_type(8)));

union Frag  { bf16x16 v; uint4 q[2]; u16 s[16]; };
union St16  { uint4 q[2]; u16 s[16]; };
union St8   { uint4 q;    u16 s[8];  };

// LDS row stride: 56 u16 = 112 B. 16B-aligned (112 = 7*16) and 28*m mod 64
// is a permutation over 16 consecutive rows -> conflict-free ds_load_b128
// fragment reads.
#define LDSP 56

// ---------------------------------------------------------------------------
// Generic strided/batched bf16 WMMA GEMM, double-buffered software pipeline:
//   C[z][m,n] = epilogue( sum_k A[z][m,k] * B[z][k,n] )
// Tile 128x64x32, 256 threads = 8 wave32 waves; each wave owns a 32x32
// sub-tile -> 4 v_wmma_f32_16x16x32_bf16 per k-step on 8 ds_load_b128 from
// fragment-major LDS tiles As[m][k], Bs[n][k].
// Pipeline: stage next tile's global loads into registers while WMMAs consume
// the current LDS buffer; commit regs -> alternate buffer; ONE barrier/step.
// Fill paths: vec-k (b128/b128), vec-m|n (b128 + 8x ds_store_b16), scalar
// bounds-checked fallback only for edge tiles.
// Epilogue: (+bias[n]) * scale, optional exact GELU, optional *rowscale[m],
// bf16 or f32 output, optional f32 accumulate-into-C (residual add).
// ---------------------------------------------------------------------------
__global__ __launch_bounds__(256) void k_wmma_gemm(
    const u16* __restrict__ A, long long a_rs, long long a_cs,
    long long a_z0, long long a_z1,
    const u16* __restrict__ Bw, long long b_rs, long long b_cs,
    long long b_z0, long long b_z1,
    void* __restrict__ Cp, long long c_rs, long long c_cs,
    long long c_z0, long long c_z1,
    const float* __restrict__ bias,
    const float* __restrict__ rowscale, long long r_z0, long long r_z1,
    int Mdim, int Ndim, int Kdim, int zdim1,
    float scale, int act, int c_bf16, int c_accum)
{
    __shared__ u16 As[2][128][LDSP];   // [buf][m][k]
    __shared__ u16 Bs[2][64][LDSP];    // [buf][n][k]

    const int tid = threadIdx.x;
    const int z   = blockIdx.z;
    const int z0  = z / zdim1;
    const int z1  = z - z0 * zdim1;

    A  += (long long)z0 * a_z0 + (long long)z1 * a_z1;
    Bw += (long long)z0 * b_z0 + (long long)z1 * b_z1;
    const float* rsc = rowscale ? rowscale + (long long)z0 * r_z0 + (long long)z1 * r_z1
                                : nullptr;
    const long long cbase = (long long)z0 * c_z0 + (long long)z1 * c_z1;

    const int tileM = blockIdx.y * 128;
    const int tileN = blockIdx.x * 64;

    const int wave = tid >> 5;
    const int lane = tid & 31;
    const int wm   = (wave >> 1) * 32;
    const int wn   = (wave & 1) * 32;
    const int half = lane >> 4;
    const int kh   = half * 8;
    const int l16  = lane & 15;

    f32x8 acc00 = {0.f,0.f,0.f,0.f,0.f,0.f,0.f,0.f};
    f32x8 acc01 = acc00, acc10 = acc00, acc11 = acc00;

    const bool a_vk = (a_cs == 1);
    const bool a_vm = (a_rs == 1);
    const bool b_vk = (b_rs == 1);
    const bool b_vn = (b_cs == 1);

    St16 aSt; St8 bSt;
    int aPath = 2, bPath = 2;

    // ---- stage: issue global loads for k-step at kb into registers ----
    auto stage_load = [&](int kb) {
        const bool fullK = (kb + 32 <= Kdim);
        const bool fullA = fullK && (tileM + 128 <= Mdim);
        const bool fullB = fullK && (tileN + 64 <= Ndim);
        if (fullA && a_vk) {
            aPath = 0;
            int m = tid >> 2, kk = (tid & 3) * 8;
            aSt.q[0] = *(const uint4*)(A + (long long)(tileM + m)      * a_rs + (kb + kk));
            aSt.q[1] = *(const uint4*)(A + (long long)(tileM + m + 64) * a_rs + (kb + kk));
        } else if (fullA && a_vm) {
            aPath = 1;
            int k = tid >> 4, mm = (tid & 15) * 8;
            aSt.q[0] = *(const uint4*)(A + (long long)(kb + k)      * a_cs + (tileM + mm));
            aSt.q[1] = *(const uint4*)(A + (long long)(kb + k + 16) * a_cs + (tileM + mm));
        } else {
            aPath = 2;
#pragma unroll
            for (int j = 0; j < 16; ++j) {
                int idx = tid + j * 256;
                int m = idx >> 5, k = idx & 31;
                int gm = tileM + m, gk = kb + k;
                aSt.s[j] = (gm < Mdim && gk < Kdim)
                         ? A[(long long)gm * a_rs + (long long)gk * a_cs] : (u16)0;
            }
        }
        if (fullB && b_vk) {
            bPath = 0;
            int n = tid >> 2, kk = (tid & 3) * 8;
            bSt.q = *(const uint4*)(Bw + (long long)(tileN + n) * b_cs + (kb + kk));
        } else if (fullB && b_vn) {
            bPath = 1;
            int k = tid >> 3, nn = (tid & 7) * 8;
            bSt.q = *(const uint4*)(Bw + (long long)(kb + k) * b_rs + (tileN + nn));
        } else {
            bPath = 2;
#pragma unroll
            for (int j = 0; j < 8; ++j) {
                int idx = tid + j * 256;
                int k = idx >> 6, n = idx & 63;
                int gk = kb + k, gn = tileN + n;
                bSt.s[j] = (gk < Kdim && gn < Ndim)
                         ? Bw[(long long)gk * b_rs + (long long)gn * b_cs] : (u16)0;
            }
        }
    };

    // ---- commit staged registers into LDS buffer ----
    auto stage_store = [&](int buf) {
        if (aPath == 0) {
            int m = tid >> 2, kk = (tid & 3) * 8;
            *(uint4*)&As[buf][m][kk]      = aSt.q[0];
            *(uint4*)&As[buf][m + 64][kk] = aSt.q[1];
        } else if (aPath == 1) {
            int k = tid >> 4, mm = (tid & 15) * 8;
#pragma unroll
            for (int j = 0; j < 8; ++j) {
                As[buf][mm + j][k]      = aSt.s[j];
                As[buf][mm + j][k + 16] = aSt.s[8 + j];
            }
        } else {
#pragma unroll
            for (int j = 0; j < 16; ++j) {
                int idx = tid + j * 256;
                As[buf][idx >> 5][idx & 31] = aSt.s[j];
            }
        }
        if (bPath == 0) {
            int n = tid >> 2, kk = (tid & 3) * 8;
            *(uint4*)&Bs[buf][n][kk] = bSt.q;
        } else if (bPath == 1) {
            int k = tid >> 3, nn = (tid & 7) * 8;
#pragma unroll
            for (int j = 0; j < 8; ++j) Bs[buf][nn + j][k] = bSt.s[j];
        } else {
#pragma unroll
            for (int j = 0; j < 8; ++j) {
                int idx = tid + j * 256;
                Bs[buf][idx & 63][idx >> 6] = bSt.s[j];
            }
        }
    };

    const int nsteps = (Kdim + 31) / 32;

    // prologue: fill buffer 0
    stage_load(0);
    stage_store(0);
    __syncthreads();

    int cur = 0;
    for (int s = 0; s < nsteps; ++s) {
        const bool have_next = (s + 1 < nsteps);
        if (have_next) stage_load((s + 1) * 32);   // overlaps with WMMAs below

        Frag a0, a1, b0, b1;
        a0.q[0] = *(const uint4*)&As[cur][wm + l16][kh];
        a0.q[1] = *(const uint4*)&As[cur][wm + l16][kh + 16];
        a1.q[0] = *(const uint4*)&As[cur][wm + 16 + l16][kh];
        a1.q[1] = *(const uint4*)&As[cur][wm + 16 + l16][kh + 16];
        b0.q[0] = *(const uint4*)&Bs[cur][wn + l16][kh];
        b0.q[1] = *(const uint4*)&Bs[cur][wn + l16][kh + 16];
        b1.q[0] = *(const uint4*)&Bs[cur][wn + 16 + l16][kh];
        b1.q[1] = *(const uint4*)&Bs[cur][wn + 16 + l16][kh + 16];

        acc00 = __builtin_amdgcn_wmma_f32_16x16x32_bf16(
                    false, a0.v, false, b0.v, (short)0, acc00, false, false);
        acc01 = __builtin_amdgcn_wmma_f32_16x16x32_bf16(
                    false, a0.v, false, b1.v, (short)0, acc01, false, false);
        acc10 = __builtin_amdgcn_wmma_f32_16x16x32_bf16(
                    false, a1.v, false, b0.v, (short)0, acc10, false, false);
        acc11 = __builtin_amdgcn_wmma_f32_16x16x32_bf16(
                    false, a1.v, false, b1.v, (short)0, acc11, false, false);

        if (have_next) {
            stage_store(cur ^ 1);   // other buffer: no race with readers of cur
            __syncthreads();        // single barrier per k-step
            cur ^= 1;
        }
    }

    // ---- epilogue; D layout: VGPR r, lane l -> M = r + 8*(l>>4), N = l&15 ----
#pragma unroll
    for (int i = 0; i < 2; ++i) {
#pragma unroll
        for (int t = 0; t < 2; ++t) {
            f32x8 av = i ? (t ? acc11 : acc10) : (t ? acc01 : acc00);
#pragma unroll
            for (int r = 0; r < 8; ++r) {
                int gm = tileM + wm + i * 16 + r + 8 * half;
                int gn = tileN + wn + t * 16 + l16;
                if (gm >= Mdim || gn >= Ndim) continue;
                float val = av[r];
                if (bias) val += bias[gn];
                val *= scale;
                if (act == 1) val = gelu_exact(val);
                if (rsc) val *= rsc[gm];
                long long cidx = cbase + (long long)gm * c_rs + (long long)gn * c_cs;
                if (c_bf16) {
                    ((u16*)Cp)[cidx] = f2bf(val);
                } else {
                    float* Cf = (float*)Cp;
                    Cf[cidx] = c_accum ? (Cf[cidx] + val) : val;
                }
            }
        }
    }
}

// ---------------------------------------------------------------------------
// Elementwise / reduction kernels
// ---------------------------------------------------------------------------
__global__ void k_f32_to_bf16(const float* __restrict__ in, u16* __restrict__ out,
                              long long n) {
    long long i = (long long)blockIdx.x * blockDim.x + threadIdx.x;
    if (i < n) out[i] = f2bf(in[i]);
}

__global__ void k_embed(const float* __restrict__ expr, const int* __restrict__ gene_ids,
                        const float* __restrict__ gene_table, const float* __restrict__ pos_table,
                        const float* __restrict__ value_table, float* __restrict__ x) {
    long long i = (long long)blockIdx.x * blockDim.x + threadIdx.x;
    long long total = (long long)Bsz * Kg * Dm;
    if (i >= total) return;
    int d = (int)(i % Dm);
    long long r = i / Dm;
    int j = (int)(r % Kg);
    int b = (int)(r / Kg);
    int bin = (int)(expr[(long long)b * Kg + j] * NBb);
    bin = bin < 0 ? 0 : (bin > NBb - 1 ? NBb - 1 : bin);
    float v = gene_table[(long long)gene_ids[j] * Dm + d]
            + pos_table[(long long)j * Dm + d]
            + value_table[(long long)bin * Dm + d];
    x[((long long)b * Ntok + 1 + j) * Dm + d] = v;
}

__global__ __launch_bounds__(256) void k_spatial(
    const float* __restrict__ coords, const float* __restrict__ sp_w,
    const float* __restrict__ sp_b, const float* __restrict__ g,
    const float* __restrict__ bb, float* __restrict__ x) {
    __shared__ float sh[Dm];
    __shared__ float red[256];
    __shared__ float s_mu, s_rstd;
    int b = blockIdx.x, tid = threadIdx.x;
    float c0 = coords[b * 2 + 0], c1 = coords[b * 2 + 1];
    for (int d = tid; d < Dm; d += 256)
        sh[d] = c0 * sp_w[d] + c1 * sp_w[Dm + d] + sp_b[d];
    __syncthreads();
    float s = 0.f, s2 = 0.f;
    for (int d = tid; d < Dm; d += 256) { float v = sh[d]; s += v; s2 += v * v; }
    red[tid] = s; __syncthreads();
    for (int o = 128; o > 0; o >>= 1) { if (tid < o) red[tid] += red[tid + o]; __syncthreads(); }
    float mu = red[0] / Dm; __syncthreads();
    red[tid] = s2; __syncthreads();
    for (int o = 128; o > 0; o >>= 1) { if (tid < o) red[tid] += red[tid + o]; __syncthreads(); }
    if (tid == 0) { s_mu = mu; s_rstd = rsqrtf(red[0] / Dm - mu * mu + 1e-5f); }
    __syncthreads();
    float rstd = s_rstd; mu = s_mu;
    for (int d = tid; d < Dm; d += 256) {
        float v = (sh[d] - mu) * rstd * g[d] + bb[d];
        x[((long long)b * Ntok) * Dm + d] = gelu_exact(v);
    }
}

__global__ __launch_bounds__(256) void k_layernorm_bf16(
    const float* __restrict__ X, const float* __restrict__ g,
    const float* __restrict__ bb, u16* __restrict__ out, int Dd) {
    __shared__ float red[256];
    __shared__ float s_mu, s_rstd;
    long long row = blockIdx.x;
    const float* xr = X + row * Dd;
    u16* orow = out + row * Dd;
    int tid = threadIdx.x;
    float s = 0.f, s2 = 0.f;
    for (int d = tid; d < Dd; d += 256) { float v = xr[d]; s += v; s2 += v * v; }
    red[tid] = s; __syncthreads();
    for (int o = 128; o > 0; o >>= 1) { if (tid < o) red[tid] += red[tid + o]; __syncthreads(); }
    float mu = red[0] / Dd; __syncthreads();
    red[tid] = s2; __syncthreads();
    for (int o = 128; o > 0; o >>= 1) { if (tid < o) red[tid] += red[tid + o]; __syncthreads(); }
    if (tid == 0) { s_mu = mu; s_rstd = rsqrtf(red[0] / Dd - mu * mu + 1e-5f); }
    __syncthreads();
    float rstd = s_rstd; mu = s_mu;
    for (int d = tid; d < Dd; d += 256)
        orow[d] = f2bf((xr[d] - mu) * rstd * g[d] + bb[d]);
}

__global__ void k_diag(const u16* __restrict__ q, float* __restrict__ diag) {
    long long i = (long long)blockIdx.x * blockDim.x + threadIdx.x;
    long long total = (long long)Bsz * Hh * Ntok;
    if (i >= total) return;
    int n = (int)(i % Ntok);
    long long r = i / Ntok;
    int h = (int)(r % Hh);
    int b = (int)(r / Hh);
    const u16* base = q + ((long long)b * Ntok + n) * Dm + h * DH;
    float s = 0.f;
    for (int d = 0; d < DH; ++d) { float v = bf2f(base[d]); s += v * v; }
    diag[i] = 0.5f * s;
}

__global__ __launch_bounds__(256) void k_rowmax(const float* __restrict__ dd,
                                                float* __restrict__ rowmax) {
    __shared__ float red[256];
    long long row = blockIdx.x;
    int tid = threadIdx.x;
    float m = -3.4e38f;
    for (int c = tid; c < Mf; c += 256) m = fmaxf(m, dd[row * Mf + c]);
    red[tid] = m; __syncthreads();
    for (int o = 128; o > 0; o >>= 1) { if (tid < o) red[tid] = fmaxf(red[tid], red[tid + o]); __syncthreads(); }
    if (tid == 0) rowmax[row] = red[0];
}

__global__ __launch_bounds__(256) void k_global_max(const float* __restrict__ in,
                                                    float* __restrict__ out, long long n) {
    __shared__ float red[256];
    int tid = threadIdx.x;
    float m = -3.4e38f;
    for (long long i = tid; i < n; i += 256) m = fmaxf(m, in[i]);
    red[tid] = m; __syncthreads();
    for (int o = 128; o > 0; o >>= 1) { if (tid < o) red[tid] = fmaxf(red[tid], red[tid + o]); __syncthreads(); }
    if (tid == 0) out[0] = red[0];
}

__global__ void k_favor_exp(const float* __restrict__ dd, const float* __restrict__ diag,
                            const float* __restrict__ mx, int mx_scalar,
                            u16* __restrict__ out) {
    long long i = (long long)blockIdx.x * blockDim.x + threadIdx.x;
    long long total = (long long)Bsz * Hh * Ntok * Mf;
    if (i >= total) return;
    long long row = i / Mf;
    float m = mx_scalar ? mx[0] : mx[row];
    out[i] = f2bf(0.0625f * (expf(dd[i] - diag[row] - m) + 1e-4f));
}

__global__ __launch_bounds__(256) void k_ksum(const u16* __restrict__ kp,
                                              float* __restrict__ ksum) {
    int z = blockIdx.x, m = threadIdx.x;
    float s = 0.f;
    const u16* base = kp + (long long)z * Ntok * Mf + m;
    for (int n = 0; n < Ntok; ++n) s += bf2f(base[(long long)n * Mf]);
    ksum[(long long)z * Mf + m] = s;
}

__global__ void k_dinv(const u16* __restrict__ qp, const float* __restrict__ ksum,
                       float* __restrict__ dinv) {
    long long r = (long long)blockIdx.x * blockDim.x + threadIdx.x;
    long long total = (long long)Bsz * Hh * Ntok;
    if (r >= total) return;
    long long z = r / Ntok;
    const u16* qr = qp + r * Mf;
    const float* ks = ksum + z * Mf;
    float s = 0.f;
    for (int m = 0; m < Mf; ++m) s += bf2f(qr[m]) * ks[m];
    dinv[r] = 1.0f / s;
}

__global__ void k_scores(const float* __restrict__ qc, const float* __restrict__ kc,
                         float* __restrict__ s) {
    long long i = (long long)blockIdx.x * blockDim.x + threadIdx.x;
    if (i >= (long long)Bsz * Kg) return;
    int b = (int)(i / Kg);
    const float* q = qc + (long long)b * Dm;
    const float* k = kc + i * Dm;
    float acc = 0.f;
    for (int d = 0; d < Dm; ++d) acc += q[d] * k[d];
    s[i] = acc * 0.044194173824159216f;   // 1/sqrt(512)
}

__global__ __launch_bounds__(256) void k_softmax(float* __restrict__ s) {
    __shared__ float red[256];
    int b = blockIdx.x, tid = threadIdx.x;
    float* row = s + (long long)b * Kg;
    float m = -3.4e38f;
    for (int k = tid; k < Kg; k += 256) m = fmaxf(m, row[k]);
    red[tid] = m; __syncthreads();
    for (int o = 128; o > 0; o >>= 1) { if (tid < o) red[tid] = fmaxf(red[tid], red[tid + o]); __syncthreads(); }
    m = red[0]; __syncthreads();
    float ssum = 0.f;
    for (int k = tid; k < Kg; k += 256) { float e = expf(row[k] - m); row[k] = e; ssum += e; }
    red[tid] = ssum; __syncthreads();
    for (int o = 128; o > 0; o >>= 1) { if (tid < o) red[tid] += red[tid + o]; __syncthreads(); }
    float inv = 1.0f / red[0];
    for (int k = tid; k < Kg; k += 256) row[k] *= inv;
}

__global__ __launch_bounds__(256) void k_pooled(const float* __restrict__ attn,
                                                const float* __restrict__ vc,
                                                float* __restrict__ pooled) {
    int b = blockIdx.x, tid = threadIdx.x;
    for (int d = tid; d < Dm; d += 256) {
        float acc = 0.f;
        for (int k = 0; k < Kg; ++k)
            acc += attn[(long long)b * Kg + k] * vc[((long long)b * Kg + k) * Dm + d];
        pooled[(long long)b * Dm + d] = acc;
    }
}

__global__ void k_small_gemm(const float* __restrict__ A, const float* __restrict__ W,
                             const float* __restrict__ bias, float* __restrict__ out,
                             int Mm, int Kk, int Nn) {
    long long i = (long long)blockIdx.x * blockDim.x + threadIdx.x;
    if (i >= (long long)Mm * Nn) return;
    int n = (int)(i % Nn), m = (int)(i / Nn);
    float acc = bias[n];
    for (int k = 0; k < Kk; ++k) acc += A[(long long)m * Kk + k] * W[(long long)k * Nn + n];
    out[i] = acc;
}

// ---------------------------------------------------------------------------
// Host side
// ---------------------------------------------------------------------------
extern "C" void kernel_launch(void* const* d_in, const int* in_sizes, int n_in,
                              void* d_out, int out_size, void* d_ws, size_t ws_size,
                              hipStream_t stream) {
    const float* expr        = (const float*)d_in[0];
    const float* coords      = (const float*)d_in[1];
    const int*   gene_ids    = (const int*)  d_in[2];
    const float* gene_table  = (const float*)d_in[3];
    const float* pos_table   = (const float*)d_in[4];
    const float* value_table = (const float*)d_in[5];
    const float* sp_w   = (const float*)d_in[6];
    const float* sp_b   = (const float*)d_in[7];
    const float* sp_ln_g= (const float*)d_in[8];
    const float* sp_ln_b= (const float*)d_in[9];
    const float* ln1_g  = (const float*)d_in[10];
    const float* ln1_b  = (const float*)d_in[11];
    const float* qw = (const float*)d_in[12]; const float* qb = (const float*)d_in[13];
    const float* kw = (const float*)d_in[14]; const float* kb = (const float*)d_in[15];
    const float* vw = (const float*)d_in[16]; const float* vb = (const float*)d_in[17];
    const float* ow = (const float*)d_in[18]; const float* ob = (const float*)d_in[19];
    const float* proj  = (const float*)d_in[20];
    const float* ln2_g = (const float*)d_in[21];
    const float* ln2_b = (const float*)d_in[22];
    const float* ff1_w = (const float*)d_in[23]; const float* ff1_b = (const float*)d_in[24];
    const float* ff2_w = (const float*)d_in[25]; const float* ff2_b = (const float*)d_in[26];
    const float* cq_w = (const float*)d_in[27]; const float* cq_b = (const float*)d_in[28];
    const float* ck_w = (const float*)d_in[29]; const float* ck_b = (const float*)d_in[30];
    const float* cv_w = (const float*)d_in[31]; const float* cv_b = (const float*)d_in[32];
    const float* co_w = (const float*)d_in[33]; const float* co_b = (const float*)d_in[34];

    const long long BN  = (long long)Bsz * Ntok;           // 65568 rows
    const long long BHN = (long long)Bsz * Hh * Ntok;

    // ---- workspace carve (deterministic) ----
    char* wsb = (char*)d_ws;
    size_t off = 0;
    auto carve = [&](size_t bytes) -> void* {
        void* p = wsb + off;
        off = (off + bytes + 255) & ~(size_t)255;
        return p;
    };
    float* x    = (float*)carve(BN * Dm * 4);             // master activations
    u16*   hbf  = (u16*)  carve(BN * Dm * 2);             // LN output / x as bf16
    u16*   qbf  = (u16*)  carve(BN * Dm * 2);             // also reused as o_bf
    u16*   kbf  = (u16*)  carve(BN * Dm * 2);
    u16*   vbf  = (u16*)  carve(BN * Dm * 2);
    float* dd   = (float*)carve(BHN * Mf * 4);            // reused: ffmid / kc,vc
    u16*   qp   = (u16*)  carve(BHN * Mf * 2);
    u16*   kp   = (u16*)  carve(BHN * Mf * 2);
    float* diag = (float*)carve(BHN * 4);
    float* rmax = (float*)carve(BHN * 4);
    float* mx1  = (float*)carve(256);
    float* ksum = (float*)carve((long long)Bsz * Hh * Mf * 4);
    float* dinv = (float*)carve(BHN * 4);
    u16*   ctx  = (u16*)  carve((long long)Bsz * Hh * Mf * DH * 2);
    float* qc     = (float*)carve((long long)Bsz * Dm * 4);
    float* scores = (float*)carve((long long)Bsz * Kg * 4);
    float* pooled = (float*)carve((long long)Bsz * Dm * 4);
    // bf16 weights
    u16* qw_bf  = (u16*)carve((long long)Ll * Dm * Dm * 2);
    u16* kw_bf  = (u16*)carve((long long)Ll * Dm * Dm * 2);
    u16* vw_bf  = (u16*)carve((long long)Ll * Dm * Dm * 2);
    u16* ow_bf  = (u16*)carve((long long)Ll * Dm * Dm * 2);
    u16* ff1_bf = (u16*)carve((long long)Ll * Dm * FF * 2);
    u16* ff2_bf = (u16*)carve((long long)Ll * FF * Dm * 2);
    u16* pj_bf  = (u16*)carve((long long)Ll * Mf * DH * 2);
    u16* cq_bf  = (u16*)carve((long long)Dm * Dm * 2);
    u16* ck_bf  = (u16*)carve((long long)Dm * Dm * 2);
    u16* cv_bf  = (u16*)carve((long long)Dm * Dm * 2);
    // aliases into the big dd region
    u16*   ffmid = (u16*)dd;                               // B*N*FF bf16
    float* kc    = (float*)dd;                             // B*K*D f32
    float* vc    = kc + (long long)Bsz * Kg * Dm;

    auto cvt = [&](const float* src, u16* dst, long long n) {
        k_f32_to_bf16<<<dim3((unsigned)((n + 255) / 256)), 256, 0, stream>>>(src, dst, n);
    };
    cvt(qw, qw_bf,  (long long)Ll * Dm * Dm);
    cvt(kw, kw_bf,  (long long)Ll * Dm * Dm);
    cvt(vw, vw_bf,  (long long)Ll * Dm * Dm);
    cvt(ow, ow_bf,  (long long)Ll * Dm * Dm);
    cvt(ff1_w, ff1_bf, (long long)Ll * Dm * FF);
    cvt(ff2_w, ff2_bf, (long long)Ll * FF * Dm);
    cvt(proj, pj_bf, (long long)Ll * Mf * DH);
    cvt(cq_w, cq_bf, (long long)Dm * Dm);
    cvt(ck_w, ck_bf, (long long)Dm * Dm);
    cvt(cv_w, cv_bf, (long long)Dm * Dm);

    auto gemm = [&](const u16* A, long long a_rs, long long a_cs, long long a_z0, long long a_z1,
                    const u16* Bm, long long b_rs, long long b_cs, long long b_z0, long long b_z1,
                    void* C, long long c_rs, long long c_cs, long long c_z0, long long c_z1,
                    const float* bias, const float* rowscale, long long r_z0, long long r_z1,
                    int Md, int Nd, int Kd, int z0n, int z1n,
                    float scale, int act, int cbf, int cacc) {
        dim3 g((unsigned)((Nd + 63) / 64), (unsigned)((Md + 127) / 128),
               (unsigned)(z0n * z1n));
        k_wmma_gemm<<<g, 256, 0, stream>>>(A, a_rs, a_cs, a_z0, a_z1,
                                           Bm, b_rs, b_cs, b_z0, b_z1,
                                           C, c_rs, c_cs, c_z0, c_z1,
                                           bias, rowscale, r_z0, r_z1,
                                           Md, Nd, Kd, z1n, scale, act, cbf, cacc);
    };

    const float dn = 0.35355339059327373f;   // DH^-0.25 = 64^-0.25

    // ---- embedding + spatial token ----
    {
        long long n = (long long)Bsz * Kg * Dm;
        k_embed<<<dim3((unsigned)((n + 255) / 256)), 256, 0, stream>>>(
            expr, gene_ids, gene_table, pos_table, value_table, x);
        k_spatial<<<Bsz, 256, 0, stream>>>(coords, sp_w, sp_b, sp_ln_g, sp_ln_b, x);
    }

    for (int l = 0; l < Ll; ++l) {
        const long long wD2 = (long long)l * Dm * Dm;
        // LN1 -> hbf
        k_layernorm_bf16<<<(unsigned)BN, 256, 0, stream>>>(x, ln1_g + l * Dm, ln1_b + l * Dm, hbf, Dm);
        // Q/K (scaled by d^-0.25), V projections (bf16 out)
        gemm(hbf, Dm, 1, 0, 0, qw_bf + wD2, Dm, 1, 0, 0, qbf, Dm, 1, 0, 0,
             qb + l * Dm, nullptr, 0, 0, (int)BN, Dm, Dm, 1, 1, dn, 0, 1, 0);
        gemm(hbf, Dm, 1, 0, 0, kw_bf + wD2, Dm, 1, 0, 0, kbf, Dm, 1, 0, 0,
             kb + l * Dm, nullptr, 0, 0, (int)BN, Dm, Dm, 1, 1, dn, 0, 1, 0);
        gemm(hbf, Dm, 1, 0, 0, vw_bf + wD2, Dm, 1, 0, 0, vbf, Dm, 1, 0, 0,
             vb + l * Dm, nullptr, 0, 0, (int)BN, Dm, Dm, 1, 1, 1.0f, 0, 1, 0);

        // ---- FAVOR+ features for Q ----
        k_diag<<<(unsigned)((BHN + 255) / 256), 256, 0, stream>>>(qbf, diag);
        // dd[z][n,m] = q[b,n,h,:] . proj[m,:]   (z = b*H+h)
        gemm(qbf, Dm, 1, (long long)Ntok * Dm, DH,
             pj_bf + (long long)l * Mf * DH, 1, DH, 0, 0,
             dd, Mf, 1, (long long)Hh * Ntok * Mf, (long long)Ntok * Mf,
             nullptr, nullptr, 0, 0, Ntok, Mf, DH, Bsz, Hh, 1.0f, 0, 0, 0);
        k_rowmax<<<(unsigned)BHN, 256, 0, stream>>>(dd, rmax);
        {
            long long n = BHN * Mf;
            k_favor_exp<<<(unsigned)((n + 255) / 256), 256, 0, stream>>>(dd, diag, rmax, 0, qp);
        }
        // ---- FAVOR+ features for K (global max) ----
        k_diag<<<(unsigned)((BHN + 255) / 256), 256, 0, stream>>>(kbf, diag);
        gemm(kbf, Dm, 1, (long long)Ntok * Dm, DH,
             pj_bf + (long long)l * Mf * DH, 1, DH, 0, 0,
             dd, Mf, 1, (long long)Hh * Ntok * Mf, (long long)Ntok * Mf,
             nullptr, nullptr, 0, 0, Ntok, Mf, DH, Bsz, Hh, 1.0f, 0, 0, 0);
        k_rowmax<<<(unsigned)BHN, 256, 0, stream>>>(dd, rmax);
        k_global_max<<<1, 256, 0, stream>>>(rmax, mx1, BHN);
        {
            long long n = BHN * Mf;
            k_favor_exp<<<(unsigned)((n + 255) / 256), 256, 0, stream>>>(dd, diag, mx1, 1, kp);
        }

        // dinv
        k_ksum<<<Bsz * Hh, Mf, 0, stream>>>(kp, ksum);
        k_dinv<<<(unsigned)((BHN + 255) / 256), 256, 0, stream>>>(qp, ksum, dinv);

        // ctx[z][m,d] = sum_n kp[z,n,m] * v[b,n,h,d]
        gemm(kp, 1, Mf, (long long)Hh * Ntok * Mf, (long long)Ntok * Mf,
             vbf, Dm, 1, (long long)Ntok * Dm, DH,
             ctx, DH, 1, (long long)Hh * Mf * DH, (long long)Mf * DH,
             nullptr, nullptr, 0, 0, Mf, DH, Ntok, Bsz, Hh, 1.0f, 0, 1, 0);

        // o[b,n,h,d] = dinv[z,n] * sum_m qp[z,n,m] * ctx[z,m,d]   -> reuse qbf
        u16* obf = qbf;
        gemm(qp, Mf, 1, (long long)Hh * Ntok * Mf, (long long)Ntok * Mf,
             ctx, DH, 1, (long long)Hh * Mf * DH, (long long)Mf * DH,
             obf, Dm, 1, (long long)Ntok * Dm, DH,
             nullptr, dinv, (long long)Hh * Ntok, Ntok,
             Ntok, DH, Mf, Bsz, Hh, 1.0f, 0, 1, 0);

        // x += o @ ow + ob   (f32 accumulate into x)
        gemm(obf, Dm, 1, 0, 0, ow_bf + wD2, Dm, 1, 0, 0, x, Dm, 1, 0, 0,
             ob + l * Dm, nullptr, 0, 0, (int)BN, Dm, Dm, 1, 1, 1.0f, 0, 0, 1);

        // FFN
        k_layernorm_bf16<<<(unsigned)BN, 256, 0, stream>>>(x, ln2_g + l * Dm, ln2_b + l * Dm, hbf, Dm);
        gemm(hbf, Dm, 1, 0, 0, ff1_bf + (long long)l * Dm * FF, FF, 1, 0, 0,
             ffmid, FF, 1, 0, 0, ff1_b + (long long)l * FF, nullptr, 0, 0,
             (int)BN, FF, Dm, 1, 1, 1.0f, /*gelu*/1, 1, 0);
        gemm(ffmid, FF, 1, 0, 0, ff2_bf + (long long)l * FF * Dm, Dm, 1, 0, 0,
             x, Dm, 1, 0, 0, ff2_b + l * Dm, nullptr, 0, 0,
             (int)BN, Dm, FF, 1, 1, 1.0f, 0, 0, 1);
    }

    // ---- cross-attention pooling ----
    cvt(x, hbf, BN * Dm);   // x as bf16
    // qc[b,:] = x[b,0,:] @ cq_w + cq_b   (row stride N*D picks token 0 per batch)
    gemm(hbf, (long long)Ntok * Dm, 1, 0, 0, cq_bf, Dm, 1, 0, 0,
         qc, Dm, 1, 0, 0, cq_b, nullptr, 0, 0, Bsz, Dm, Dm, 1, 1, 1.0f, 0, 0, 0);
    // kc/vc: per-batch GEMM over tokens 1..K
    gemm(hbf + Dm, Dm, 1, (long long)Ntok * Dm, 0, ck_bf, Dm, 1, 0, 0,
         kc, Dm, 1, (long long)Kg * Dm, 0, ck_b, nullptr, 0, 0,
         Kg, Dm, Dm, Bsz, 1, 1.0f, 0, 0, 0);
    gemm(hbf + Dm, Dm, 1, (long long)Ntok * Dm, 0, cv_bf, Dm, 1, 0, 0,
         vc, Dm, 1, (long long)Kg * Dm, 0, cv_b, nullptr, 0, 0,
         Kg, Dm, Dm, Bsz, 1, 1.0f, 0, 0, 0);

    {
        long long n = (long long)Bsz * Kg;
        k_scores<<<(unsigned)((n + 255) / 256), 256, 0, stream>>>(qc, kc, scores);
    }
    k_softmax<<<Bsz, 256, 0, stream>>>(scores);
    k_pooled<<<Bsz, 256, 0, stream>>>(scores, vc, pooled);
    {
        long long n = (long long)Bsz * Dm;
        k_small_gemm<<<(unsigned)((n + 255) / 256), 256, 0, stream>>>(
            pooled, co_w, co_b, (float*)d_out, Bsz, Dm, Dm);
    }
    (void)in_sizes; (void)n_in; (void)out_size; (void)ws_size;
}
